// _GATv2Block_48455821033918
// MI455X (gfx1250) — matrix operand order repaired and checked by reading, not compiled, not run on hardware
//
#include <hip/hip_runtime.h>
#include <hip/hip_bf16.h>

// ---------------------------------------------------------------------------
// GATv2 block for MI455X (gfx1250, wave32).
//  - Node GEMM x@[W_l|W_r]: v_wmma_f32_16x16x32_bf16, weights staged in LDS
//    (bf16, padded column-major) -> 64 WMMAs/wave, A fragments reused across
//    16 column tiles, B served by ds_load_b128. Store guards hoisted to a
//    wave-uniform branch (only the last of 391 blocks is partial).
//  - Edge projection (E x 16 @ 16 x 128) fused into edge kernel (avoids a
//    409MB intermediate that would spill the 192MB L2; working set ~91MB
//    stays L2-resident so edge gathers run at L2 bandwidth).
//  - Segment softmax via ordered-uint atomicMax + float atomicAdd.
//  - HEADS*OUT_CH = 4*32: one head == one wave32 -> shuffle reduction.
// ---------------------------------------------------------------------------

typedef __attribute__((ext_vector_type(16))) __bf16 v16bf;
typedef __attribute__((ext_vector_type(8)))  float  v8f;

#define HEADS 4
#define OUTT 128          // HEADS * OUT_CH == IN_CH
#define INCH 128
#define EDGED 16
#define NEG_SLOPE 0.2f
#define BN_EPS 1e-5f
#define WPAD 136          // LDS column stride in bf16 (272B: 16B-aligned, low-conflict)

__device__ __forceinline__ unsigned f2ord(float f) {
  unsigned u = __float_as_uint(f);
  return (u & 0x80000000u) ? ~u : (u | 0x80000000u);
}
__device__ __forceinline__ float ord2f(unsigned u) {
  u = (u & 0x80000000u) ? (u & 0x7FFFFFFFu) : ~u;
  return __uint_as_float(u);
}

// ---------------------------------------------------------------------------
// 0) Workspace init (workspace is poisoned by the harness each run)
// ---------------------------------------------------------------------------
__global__ void init_ws_kernel(unsigned* __restrict__ smax,
                               float* __restrict__ denom,
                               float* __restrict__ agg,
                               float* __restrict__ stats, int Ncnt) {
  const size_t nAgg = (size_t)Ncnt * OUTT;
  const size_t nSeg = (size_t)Ncnt * HEADS;
  const size_t stride = (size_t)gridDim.x * blockDim.x;
  for (size_t t = (size_t)blockIdx.x * blockDim.x + threadIdx.x; t < nAgg; t += stride) {
    agg[t] = 0.0f;
    if (t < nSeg) { smax[t] = 0x00800000u /* ordered(-FLT_MAX) */; denom[t] = 0.0f; }
    if (t < 2 * OUTT) stats[t] = 0.0f;
  }
}

// ---------------------------------------------------------------------------
// 1) Node GEMM: xlxr[N][256] = x[N][128] @ [W_l | W_r]   (WMMA bf16)
//    Block = 256 threads = 8 waves = 128 rows. Weights staged in LDS in two
//    128-col halves (34.8KB each). Each wave: 4 A fragments loaded once,
//    then 8 col tiles per half x 4 K-steps of v_wmma_f32_16x16x32_bf16.
//    Fragment packing: contiguous K = 16*half + i in both A and B (16-bit A 16x32, B 32x16, f32 C/D).
// ---------------------------------------------------------------------------
__global__ void __launch_bounds__(256)
gemm_xlxr_wmma(const float* __restrict__ x, const float* __restrict__ Wl,
               const float* __restrict__ Wr, float* __restrict__ xlxr, int Ncnt) {
  __shared__ __bf16 sW[128 * WPAD];    // 34,816 B: one 128-col half of [K][col]

  const int tid  = threadIdx.x;
  const int lane = tid & 31;
  const int wave = tid >> 5;           // 0..7
  const int half = lane >> 4;          // 0..1
  const int l15  = lane & 15;

  const bool fullBlock = (blockIdx.x * 128 + 128) <= Ncnt;   // wave-uniform

  // ---- A fragments: this wave owns rows [blockRow + wave*16, +16) ----
  const int rowA = blockIdx.x * 128 + wave * 16 + l15;
  const int arow = (rowA < Ncnt) ? rowA : (Ncnt - 1);   // clamp; stores guarded
  const float* __restrict__ xrow = x + (size_t)arow * INCH;
  v16bf afrag[4];
#pragma unroll
  for (int kk = 0; kk < 4; ++kk) {
#pragma unroll
    for (int i = 0; i < 16; ++i) {
      // A (16x32 bf16): lane(0-15)=row; elem i -> K = 16*half + i
      const int ka = kk * 32 + (half << 4) + i;
      afrag[kk][i] = (__bf16)xrow[ka];
    }
  }

  const int dRowBase = blockIdx.x * 128 + wave * 16 + half * 8;

  // ---- two halves of the fused weight matrix [W_l | W_r] ----
  for (int p = 0; p < 2; ++p) {
    const float* __restrict__ W = p ? Wr : Wl;   // [128 K][128 col] row-major
    __syncthreads();                             // LDS reuse barrier
    for (int i = tid; i < 128 * 128; i += 256) {
      const int k   = i >> 7;                    // coalesced along col
      const int col = i & 127;
      sW[col * WPAD + k] = (__bf16)W[k * 128 + col];
    }
    __syncthreads();

#pragma unroll
    for (int tn = 0; tn < 8; ++tn) {
      const int col = tn * 16 + l15;             // col within this half
      const __bf16* __restrict__ wp = sW + col * WPAD;
      v8f acc = {};
#pragma unroll
      for (int kk = 0; kk < 4; ++kk) {
        // B (32x16 bf16): lane(0-15)=col; elem i -> K = 16*half + i
        v16bf b;
        const int kb = kk * 32 + (half << 4);
#pragma unroll
        for (int i = 0; i < 16; ++i) b[i] = wp[kb + i];   // -> ds_load_b128 x2
        acc = __builtin_amdgcn_wmma_f32_16x16x32_bf16(false, afrag[kk], false, b,
                                                      (short)0, acc, false, false);
      }
      // D (16x16 f32): lanes 0-15 -> M=r, lanes 16-31 -> M=r+8; N = lane&15
      float* __restrict__ dp = xlxr + (size_t)dRowBase * 256 + p * 128 + tn * 16 + l15;
      if (fullBlock) {
        // straight-line store clause: no exec juggling on the hot path
#pragma unroll
        for (int r = 0; r < 8; ++r) dp[(size_t)r * 256] = acc[r];
      } else {
#pragma unroll
        for (int r = 0; r < 8; ++r)
          if (dRowBase + r < Ncnt) dp[(size_t)r * 256] = acc[r];
      }
    }
  }
}

// ---------------------------------------------------------------------------
// 2) Edge score: e = leaky(xl[src] + xr[dst] + ea@W_e); score = <e, att>_head
//    256 thr = 2 edges; thread c = channel; one wave == one head (32 ch).
// ---------------------------------------------------------------------------
__global__ void __launch_bounds__(256)
edge_score_kernel(const float* __restrict__ xlxr, const int* __restrict__ ei,
                  const float* __restrict__ ea, const float* __restrict__ We,
                  const float* __restrict__ att, float* __restrict__ score,
                  unsigned* __restrict__ smax, int Ecnt) {
  __shared__ float sWe[EDGED * OUTT];   // 8 KB of 320KB/WGP LDS
  __shared__ float sAtt[OUTT];
  for (int i = threadIdx.x; i < EDGED * OUTT; i += blockDim.x) sWe[i] = We[i];
  if (threadIdx.x < OUTT) sAtt[threadIdx.x] = att[threadIdx.x];
  __syncthreads();

  const int c   = threadIdx.x & 127;
  const int sub = threadIdx.x >> 7;     // which of the 2 edges in this block
  const int h   = c >> 5;

  for (long long e = (long long)blockIdx.x * 2 + sub; e < Ecnt;
       e += (long long)gridDim.x * 2) {
    const int src = ei[e];
    const int dst = ei[(size_t)Ecnt + e];
    float v = xlxr[(size_t)src * 256 + c] + xlxr[(size_t)dst * 256 + OUTT + c];
    const float* __restrict__ eaRow = ea + (size_t)e * EDGED;
#pragma unroll
    for (int d = 0; d < EDGED; ++d) v = fmaf(eaRow[d], sWe[d * OUTT + c], v);
    v = (v > 0.0f) ? v : NEG_SLOPE * v;                 // leaky_relu
    float s = v * sAtt[c];
#pragma unroll
    for (int off = 16; off > 0; off >>= 1) s += __shfl_down(s, off, 32);
    if ((threadIdx.x & 31) == 0) {
      score[(size_t)e * HEADS + h] = s;
      atomicMax(smax + (size_t)dst * HEADS + h, f2ord(s));  // segment max
    }
  }
}

// ---------------------------------------------------------------------------
// 3) exp(score - smax[dst]) and segment-sum denominators
// ---------------------------------------------------------------------------
__global__ void __launch_bounds__(256)
edge_exp_kernel(const int* __restrict__ ei, const unsigned* __restrict__ smax,
                float* __restrict__ score, float* __restrict__ denom, int Ecnt) {
  const long long idx = (long long)blockIdx.x * blockDim.x + threadIdx.x;
  const long long total = (long long)Ecnt * HEADS;
  if (idx >= total) return;
  const long long e = idx >> 2;
  const int h = (int)(idx & 3);
  const int dst = ei[(size_t)Ecnt + e];
  const float m = ord2f(smax[(size_t)dst * HEADS + h]);
  const float ex = __expf(score[idx] - m);
  score[idx] = ex;
  atomicAdd(denom + (size_t)dst * HEADS + h, ex);
}

// ---------------------------------------------------------------------------
// 4) alpha-weighted scatter aggregation: agg[dst] += xl[src] * alpha
// ---------------------------------------------------------------------------
__global__ void __launch_bounds__(256)
edge_agg_kernel(const float* __restrict__ xlxr, const int* __restrict__ ei,
                const float* __restrict__ score, const float* __restrict__ denom,
                float* __restrict__ agg, int Ecnt) {
  const int c   = threadIdx.x & 127;
  const int sub = threadIdx.x >> 7;
  const int h   = c >> 5;
  for (long long e = (long long)blockIdx.x * 2 + sub; e < Ecnt;
       e += (long long)gridDim.x * 2) {
    const int src = ei[e];
    const int dst = ei[(size_t)Ecnt + e];
    const float alpha = score[(size_t)e * HEADS + h] / denom[(size_t)dst * HEADS + h];
    atomicAdd(agg + (size_t)dst * OUTT + c, xlxr[(size_t)src * 256 + c] * alpha);
  }
}

// ---------------------------------------------------------------------------
// 5) Per-channel batch statistics (sum, sumsq) of (agg + bias)
// ---------------------------------------------------------------------------
__global__ void __launch_bounds__(128)
stats_kernel(const float* __restrict__ agg, const float* __restrict__ bias,
             float* __restrict__ stats, int Ncnt) {
  const int c = threadIdx.x;            // 128 channels
  const float b = bias[c];
  float s = 0.0f, s2 = 0.0f;
  for (int row = blockIdx.x; row < Ncnt; row += gridDim.x) {
    const float v = agg[(size_t)row * OUTT + c] + b;
    s += v; s2 += v * v;
  }
  atomicAdd(stats + c, s);
  atomicAdd(stats + OUTT + c, s2);
}

// ---------------------------------------------------------------------------
// 6) BatchNorm (batch stats) + ELU + residual skip
// ---------------------------------------------------------------------------
__global__ void __launch_bounds__(256)
finalize_kernel(const float* __restrict__ agg, const float* __restrict__ stats,
                const float* __restrict__ bias, const float* __restrict__ gamma,
                const float* __restrict__ beta, const float* __restrict__ x,
                float* __restrict__ out, int Ncnt) {
  const long long idx = (long long)blockIdx.x * blockDim.x + threadIdx.x;
  const long long total = (long long)Ncnt * OUTT;
  if (idx >= total) return;
  const int c = (int)(idx & 127);
  const float invN = 1.0f / (float)Ncnt;
  const float mu = stats[c] * invN;
  const float var = stats[OUTT + c] * invN - mu * mu;   // biased variance
  const float r = rsqrtf(var + BN_EPS);
  float v = (agg[idx] + bias[c] - mu) * r * gamma[c] + beta[c];
  v = (v > 0.0f) ? v : (__expf(v) - 1.0f);              // ELU
  out[idx] = v + x[idx];                                // skip (Identity)
}

// ---------------------------------------------------------------------------
extern "C" void kernel_launch(void* const* d_in, const int* in_sizes, int n_in,
                              void* d_out, int out_size, void* d_ws, size_t ws_size,
                              hipStream_t stream) {
  const float* x     = (const float*)d_in[0];
  const int*   ei    = (const int*)d_in[1];      // [2][E]: src row then dst row
  const float* ea    = (const float*)d_in[2];
  const float* Wl    = (const float*)d_in[3];
  const float* Wr    = (const float*)d_in[4];
  const float* We    = (const float*)d_in[5];
  const float* att   = (const float*)d_in[6];
  const float* bias  = (const float*)d_in[7];
  const float* gam   = (const float*)d_in[8];
  const float* bet   = (const float*)d_in[9];
  float* out = (float*)d_out;

  const int Ncnt = in_sizes[0] / INCH;   // 50000
  const int Ecnt = in_sizes[1] / 2;      // 800000

  // workspace layout (floats)
  float*    ws    = (float*)d_ws;
  float*    xlxr  = ws;                                       // N*256
  float*    score = xlxr + (size_t)Ncnt * 256;                // E*4
  unsigned* smax  = (unsigned*)(score + (size_t)Ecnt * HEADS);// N*4
  float*    denom = (float*)(smax + (size_t)Ncnt * HEADS);    // N*4
  float*    agg   = denom + (size_t)Ncnt * HEADS;             // N*128
  float*    stats = agg + (size_t)Ncnt * OUTT;                // 256

  // 0) init accumulators
  init_ws_kernel<<<8192, 256, 0, stream>>>(smax, denom, agg, stats, Ncnt);

  // 1) WMMA node GEMM: 128 rows per 256-thread block, LDS-resident weights
  {
    const int blocks = (Ncnt + 127) / 128;
    gemm_xlxr_wmma<<<blocks, 256, 0, stream>>>(x, Wl, Wr, xlxr, Ncnt);
  }

  // 2..4) edge phase: 2 edges per 256-thread block
  {
    const int eb = (Ecnt + 1) / 2;
    edge_score_kernel<<<eb, 256, 0, stream>>>(xlxr, ei, ea, We, att, score, smax, Ecnt);
    const long long tot = (long long)Ecnt * HEADS;
    edge_exp_kernel<<<(unsigned)((tot + 255) / 256), 256, 0, stream>>>(ei, smax, score, denom, Ecnt);
    edge_agg_kernel<<<eb, 256, 0, stream>>>(xlxr, ei, score, denom, agg, Ecnt);
  }

  // 5) batch stats, 6) BN + ELU + skip
  stats_kernel<<<2048, 128, 0, stream>>>(agg, bias, stats, Ncnt);
  {
    const long long tot = (long long)Ncnt * OUTT;
    finalize_kernel<<<(unsigned)((tot + 255) / 256), 256, 0, stream>>>(
        agg, stats, bias, gam, bet, x, out, Ncnt);
  }
}
